// HierarchicalMoE_20066087206925
// MI455X (gfx1250) — compile-verified, matrix operand order checked by reading, not run on hardware
//
#include <hip/hip_runtime.h>
#include <math.h>

// ---------------------------------------------------------------------------
// Model dims (compile-time constants from the reference)
// ---------------------------------------------------------------------------
#define HD   512           // hidden
#define IDIM 1024          // intermediate
#define BB   2             // batch
#define SS   256           // seq
#define NTOK (BB * SS)     // 512 tokens
#define RATT (BB * 4 * SS) // 2048 attention rows
#define NHEAD 8
#define HDIM  64           // head dim
#define LN_EPS 1e-5f
#define BLD  40            // padded K-stride, 32-row transposed staging
#define GBLD 72            // padded K-stride, 64-row transposed staging (GEMM)

typedef __attribute__((ext_vector_type(16))) __bf16 v16bf;
typedef __attribute__((ext_vector_type(8)))  __bf16 v8bf;
typedef __attribute__((ext_vector_type(8)))  float  v8f;

union bfpk2 { __bf16 b[2]; unsigned u; };
union bfpk4 { __bf16 b[4]; uint2 u2; };

// ---------------------------------------------------------------------------
// WMMA helpers (wave32, 16x16x32 bf16 -> f32 accumulate)
// ---------------------------------------------------------------------------
__device__ __forceinline__ v8f wmma_bf16(v16bf a, v16bf b, v8f c) {
  return __builtin_amdgcn_wmma_f32_16x16x32_bf16(false, a, false, b,
                                                 (short)0, c, false, false);
}

// Fragment with K contiguous in memory: element (idx, k) at s[idx*ld + k].
// Lane (idx, hi) holds K runs [hi*8, hi*8+8) and [16+hi*8, 16+hi*8+8):
// exactly two aligned 16-byte loads (ds_load_b128 / global_load_b128).
__device__ __forceinline__ v16bf load_frag_kc(const __bf16* s, int ld) {
  int lane = threadIdx.x & 31;
  int idx = lane & 15, hi = lane >> 4;
  const __bf16* p = s + idx * ld + hi * 8;
  union { v16bf v; v8bf h[2]; } u;
  u.h[0] = *(const v8bf*)(p);
  u.h[1] = *(const v8bf*)(p + 16);
  return u.v;
}

// ---------------------------------------------------------------------------
// Elementwise helpers
// ---------------------------------------------------------------------------
__global__ void zero_kernel(float* p, int n) {
  int i = blockIdx.x * 256 + threadIdx.x;
  if (i < n) p[i] = 0.f;
}

// ---------------------------------------------------------------------------
// Router: logits = x @ Wg^T, softmax, top-2, renormalize, scatter to lists.
// ---------------------------------------------------------------------------
__global__ void __launch_bounds__(64) router_kernel(
    const float* __restrict__ x, const float* __restrict__ Wg, int E,
    int* __restrict__ cnt, int* __restrict__ lidx, float* __restrict__ lw) {
  int tok = blockIdx.x;
  int tid = threadIdx.x;
  __shared__ float slog[64];
  float val = -1e30f;
  if (tid < E) {
    const float* xr = x + (size_t)tok * HD;
    const float* wr = Wg + (size_t)tid * HD;
    float s = 0.f;
    for (int j = 0; j < HD; j += 4) {
      float4 a = *(const float4*)&xr[j];
      float4 b = *(const float4*)&wr[j];
      s += a.x * b.x + a.y * b.y + a.z * b.z + a.w * b.w;
    }
    val = s;
  }
  slog[tid] = val;
  __syncthreads();
  if (tid == 0) {
    float mx = -1e30f;
    for (int e = 0; e < E; ++e) mx = fmaxf(mx, slog[e]);
    float sum = 0.f;
    for (int e = 0; e < E; ++e) sum += __expf(slog[e] - mx);
    int i1 = 0, i2 = 0;
    float b1v = -1e30f, b2v = -1e30f;
    for (int e = 0; e < E; ++e) {
      float sc = slog[e];
      if (sc > b1v) { b2v = b1v; i2 = i1; b1v = sc; i1 = e; }
      else if (sc > b2v) { b2v = sc; i2 = e; }
    }
    float p1 = __expf(b1v - mx) / sum;
    float p2 = __expf(b2v - mx) / sum;
    float t = p1 + p2;
    p1 /= t; p2 /= t;
    int pos1 = atomicAdd(&cnt[i1], 1);
    lidx[i1 * NTOK + pos1] = tok; lw[i1 * NTOK + pos1] = p1;
    int pos2 = atomicAdd(&cnt[i2], 1);
    lidx[i2 * NTOK + pos2] = tok; lw[i2 * NTOK + pos2] = p2;
  }
}

// ---------------------------------------------------------------------------
// Stage a 32 x 128 f32 block into transposed bf16 LDS [128][BLD].
// Row-pair processing: float4 global loads, packed b32 LDS stores.
// ---------------------------------------------------------------------------
__device__ __forceinline__ void stage_b_t(__bf16* swt, const float* src,
                                          int ldsrc, int tid) {
#pragma unroll
  for (int ii = 0; ii < 2; ++ii) {
    int i = tid + ii * 256;          // 0..511: 16 row-pairs x 32 col-groups
    int rp = i >> 5, c4 = i & 31;
    int col = c4 * 4, r = rp * 2;
    float4 w0 = *(const float4*)&src[(size_t)r * ldsrc + col];
    float4 w1 = *(const float4*)&src[(size_t)(r + 1) * ldsrc + col];
    bfpk2 p0, p1, p2, p3;
    p0.b[0] = (__bf16)w0.x; p0.b[1] = (__bf16)w1.x;
    p1.b[0] = (__bf16)w0.y; p1.b[1] = (__bf16)w1.y;
    p2.b[0] = (__bf16)w0.z; p2.b[1] = (__bf16)w1.z;
    p3.b[0] = (__bf16)w0.w; p3.b[1] = (__bf16)w1.w;
    *(unsigned*)(swt + (col + 0) * BLD + r) = p0.u;
    *(unsigned*)(swt + (col + 1) * BLD + r) = p1.u;
    *(unsigned*)(swt + (col + 2) * BLD + r) = p2.u;
    *(unsigned*)(swt + (col + 3) * BLD + r) = p3.u;
  }
}

// Stage a 64 x 128 f32 block into transposed bf16 LDS [128][GBLD].
__device__ __forceinline__ void stage_b_t64(__bf16* swt, const float* src,
                                            int ldsrc, int tid) {
#pragma unroll
  for (int ii = 0; ii < 4; ++ii) {
    int i = tid + ii * 256;          // 0..1023: 32 row-pairs x 32 col-groups
    int rp = i >> 5, c4 = i & 31;
    int col = c4 * 4, r = rp * 2;
    float4 w0 = *(const float4*)&src[(size_t)r * ldsrc + col];
    float4 w1 = *(const float4*)&src[(size_t)(r + 1) * ldsrc + col];
    bfpk2 p0, p1, p2, p3;
    p0.b[0] = (__bf16)w0.x; p0.b[1] = (__bf16)w1.x;
    p1.b[0] = (__bf16)w0.y; p1.b[1] = (__bf16)w1.y;
    p2.b[0] = (__bf16)w0.z; p2.b[1] = (__bf16)w1.z;
    p3.b[0] = (__bf16)w0.w; p3.b[1] = (__bf16)w1.w;
    *(unsigned*)(swt + (col + 0) * GBLD + r) = p0.u;
    *(unsigned*)(swt + (col + 1) * GBLD + r) = p1.u;
    *(unsigned*)(swt + (col + 2) * GBLD + r) = p2.u;
    *(unsigned*)(swt + (col + 3) * GBLD + r) = p3.u;
  }
}

// ---------------------------------------------------------------------------
// Expert FFN chain: one workgroup per (expert, 16-token tile).
//   h = gelu(x @ W1 + b1); o = h @ W2 + b2;
//   spec = LN(o @ Ws + bs) * g + be;  out[token] += w * (o + spec)
// ---------------------------------------------------------------------------
__global__ void __launch_bounds__(256) expert_ffn_kernel(
    const float* __restrict__ x,
    const float* __restrict__ W1, const float* __restrict__ b1,
    const float* __restrict__ W2, const float* __restrict__ b2,
    const float* __restrict__ Ws, const float* __restrict__ bs,
    const float* __restrict__ g,  const float* __restrict__ be,
    const int* __restrict__ cnt,
    const int* __restrict__ lidx, const float* __restrict__ lw,
    float* __restrict__ outbuf) {
  int e = blockIdx.x, tile = blockIdx.y;
  int c = cnt[e];
  if (tile * 16 >= c) return;
  int tid = threadIdx.x, lane = tid & 31, wave = tid >> 5;
  int hi = lane >> 4;

  __shared__ __bf16 sx[16 * HD];       // 16KB: x tile bf16; later o bf16
  __shared__ __bf16 sh[16 * IDIM];     // 32KB: h bf16; later spec f32 alias
  __shared__ __bf16 swt[128 * BLD];    // 10KB: transposed weight K-block
  __shared__ float  red[256], red2[256];
  __shared__ float  s_mean[16], s_rstd[16], s_wt[16];
  __shared__ int    s_tok[16];
  float* sspec = (float*)sh;           // 16 x 512 f32 alias

  if (tid < 16) {
    int idx = tile * 16 + tid;
    if (idx < c) { s_tok[tid] = lidx[e * NTOK + idx]; s_wt[tid] = lw[e * NTOK + idx]; }
    else         { s_tok[tid] = -1; s_wt[tid] = 0.f; }
  }
  __syncthreads();

  // gather x tile: float4 loads, packed 8B bf16 stores
#pragma unroll
  for (int ii = 0; ii < 8; ++ii) {
    int i = tid + ii * 256;            // 0..2047: 16 rows x 128 col-groups
    int r = i >> 7, c4 = i & 127;
    int col = c4 * 4;
    int t = s_tok[r];
    float4 xv = t >= 0 ? *(const float4*)&x[(size_t)t * HD + col]
                       : make_float4(0.f, 0.f, 0.f, 0.f);
    bfpk4 pk;
    pk.b[0] = (__bf16)xv.x; pk.b[1] = (__bf16)xv.y;
    pk.b[2] = (__bf16)xv.z; pk.b[3] = (__bf16)xv.w;
    *(uint2*)(sx + r * HD + col) = pk.u2;
  }
  __syncthreads();

  const float* W1e = W1 + (size_t)e * HD * IDIM;
  const float* b1e = b1 + (size_t)e * IDIM;
  const float* W2e = W2 + (size_t)e * IDIM * HD;
  const float* b2e = b2 + (size_t)e * HD;
  const float* Wse = Ws + (size_t)e * HD * HD;
  const float* bse = bs + (size_t)e * HD;
  const float* ge  = g  + (size_t)e * HD;
  const float* bee = be + (size_t)e * HD;

  // ---- Pass 1: h = gelu(x @ W1 + b1), 1024 cols in chunks of 128 ----
  for (int cc = 0; cc < IDIM; cc += 128) {
    v8f acc = {};
    for (int k0 = 0; k0 < HD; k0 += 32) {
      stage_b_t(swt, &W1e[(size_t)k0 * IDIM + cc], IDIM, tid);
      if (k0 + 32 < HD)
        __builtin_prefetch(&W1e[(size_t)(k0 + 32) * IDIM + cc], 0, 1);
      __syncthreads();
      v16bf a = load_frag_kc(sx + k0, HD);
      v16bf b = load_frag_kc(swt + wave * 16 * BLD, BLD);
      acc = wmma_bf16(a, b, acc);
      __syncthreads();
    }
    int n = cc + wave * 16 + (lane & 15);
    float bb = b1e[n];
#pragma unroll
    for (int v = 0; v < 8; ++v) {
      int m = v + hi * 8;
      float hv = acc[v] + bb;
      hv = 0.5f * hv * (1.0f + erff(hv * 0.70710678118f));  // exact GELU
      sh[m * IDIM + n] = (__bf16)hv;
    }
  }
  __syncthreads();

  // ---- Pass 2: o = h @ W2 + b2; f32 in regs, bf16 copy into sx ----
  v8f oreg[4];
  for (int q = 0; q < 4; ++q) {
    int cc = q * 128;
    v8f acc = {};
    for (int k0 = 0; k0 < IDIM; k0 += 32) {
      stage_b_t(swt, &W2e[(size_t)k0 * HD + cc], HD, tid);
      if (k0 + 32 < IDIM)
        __builtin_prefetch(&W2e[(size_t)(k0 + 32) * HD + cc], 0, 1);
      __syncthreads();
      v16bf a = load_frag_kc(sh + k0, IDIM);
      v16bf b = load_frag_kc(swt + wave * 16 * BLD, BLD);
      acc = wmma_bf16(a, b, acc);
      __syncthreads();
    }
    int n = cc + wave * 16 + (lane & 15);
    float bb = b2e[n];
#pragma unroll
    for (int v = 0; v < 8; ++v) acc[v] += bb;
    oreg[q] = acc;
  }
  __syncthreads();
#pragma unroll
  for (int q = 0; q < 4; ++q) {
    int n = q * 128 + wave * 16 + (lane & 15);
#pragma unroll
    for (int v = 0; v < 8; ++v) sx[(v + hi * 8) * HD + n] = (__bf16)oreg[q][v];
  }
  __syncthreads();

  // ---- Pass 3: spec_pre = o @ Ws + bs -> f32 into sspec (aliases sh) ----
  for (int q = 0; q < 4; ++q) {
    int cc = q * 128;
    v8f acc = {};
    for (int k0 = 0; k0 < HD; k0 += 32) {
      stage_b_t(swt, &Wse[(size_t)k0 * HD + cc], HD, tid);
      __syncthreads();
      v16bf a = load_frag_kc(sx + k0, HD);
      v16bf b = load_frag_kc(swt + wave * 16 * BLD, BLD);
      acc = wmma_bf16(a, b, acc);
      __syncthreads();
    }
    int n = cc + wave * 16 + (lane & 15);
    float bb = bse[n];
#pragma unroll
    for (int v = 0; v < 8; ++v) sspec[(v + hi * 8) * HD + n] = acc[v] + bb;
  }
  __syncthreads();

  // ---- LayerNorm stats: 16 threads per row, float4 reads ----
  {
    int r = tid >> 4, sub = tid & 15;     // 32 elems each
    float s = 0.f, s2 = 0.f;
#pragma unroll
    for (int j4 = 0; j4 < 8; ++j4) {
      float4 t4 = *(const float4*)&sspec[r * HD + sub * 32 + j4 * 4];
      s += t4.x + t4.y + t4.z + t4.w;
      s2 += t4.x * t4.x + t4.y * t4.y + t4.z * t4.z + t4.w * t4.w;
    }
    red[tid] = s; red2[tid] = s2;
    __syncthreads();
    if (sub == 0) {
      float ss = 0.f, ss2 = 0.f;
#pragma unroll
      for (int t2 = 0; t2 < 16; ++t2) { ss += red[(r << 4) + t2]; ss2 += red2[(r << 4) + t2]; }
      float mean = ss * (1.f / HD);
      float var = ss2 * (1.f / HD) - mean * mean;
      s_mean[r] = mean;
      s_rstd[r] = rsqrtf(var + LN_EPS);
    }
  }
  __syncthreads();

  // ---- Combine: out[token] += w * (o + LN(spec)*g + be) ----
#pragma unroll
  for (int q = 0; q < 4; ++q) {
    int n = q * 128 + wave * 16 + (lane & 15);
    float gg = ge[n], bb = bee[n];
#pragma unroll
    for (int v = 0; v < 8; ++v) {
      int m = v + hi * 8;
      int t = s_tok[m];
      if (t < 0) continue;
      float spec = (sspec[m * HD + n] - s_mean[m]) * s_rstd[m] * gg + bb;
      atomicAdd(&outbuf[(size_t)t * HD + n], s_wt[m] * (oreg[q][v] + spec));
    }
  }
}

// ---------------------------------------------------------------------------
// Generic WMMA GEMM: C[MxN] = A[MxK](f32) @ B[KxN](f32) + bias
// tile 16 x 128, K step 64 (2 WMMAs per barrier pair), 8 waves.
// MODE: 0 = f32 out, 1 = relu f32, 2 = bf16 out, 3 = bf16 transposed out.
// Requires M%16==0, N%128==0, K%64==0.
// ---------------------------------------------------------------------------
template <int MODE>
__global__ void __launch_bounds__(256) gemm_bias_kernel(
    const float* __restrict__ A, int lda,
    const float* __restrict__ Bw, int ldb,
    const float* __restrict__ bias,
    void* __restrict__ C, int ldc, int K) {
  __shared__ __bf16 sA[16 * 64];       // 2KB
  __shared__ __bf16 sBt[128 * GBLD];   // 18KB
  int tid = threadIdx.x, lane = tid & 31, wave = tid >> 5;
  int m0 = blockIdx.y * 16;
  int n0 = blockIdx.x * 128;
  v8f acc = {};
  for (int k0 = 0; k0 < K; k0 += 64) {
    {  // A: 16 rows x 16 col-groups of 4 = 256 float4 items, 1 per thread
      int r = tid >> 4, c4 = tid & 15;
      int col = c4 * 4;
      float4 av = *(const float4*)&A[(size_t)(m0 + r) * lda + k0 + col];
      bfpk4 pk;
      pk.b[0] = (__bf16)av.x; pk.b[1] = (__bf16)av.y;
      pk.b[2] = (__bf16)av.z; pk.b[3] = (__bf16)av.w;
      *(uint2*)(sA + r * 64 + col) = pk.u2;
    }
    stage_b_t64(sBt, &Bw[(size_t)k0 * ldb + n0], ldb, tid);
    if (k0 + 64 < K) __builtin_prefetch(&Bw[(size_t)(k0 + 64) * ldb + n0], 0, 1);
    __syncthreads();
    v16bf a0 = load_frag_kc(sA, 64);
    v16bf b0 = load_frag_kc(sBt + wave * 16 * GBLD, GBLD);
    acc = wmma_bf16(a0, b0, acc);
    v16bf a1 = load_frag_kc(sA + 32, 64);
    v16bf b1 = load_frag_kc(sBt + wave * 16 * GBLD + 32, GBLD);
    acc = wmma_bf16(a1, b1, acc);
    __syncthreads();
  }
  int n = n0 + wave * 16 + (lane & 15);
  int hi = lane >> 4;
  float bv = bias ? bias[n] : 0.f;
#pragma unroll
  for (int v = 0; v < 8; ++v) {
    int m = m0 + v + hi * 8;
    float r = acc[v] + bv;
    if (MODE == 1) r = fmaxf(r, 0.f);
    if (MODE == 3)      ((__bf16*)C)[(size_t)n * ldc + m] = (__bf16)r;
    else if (MODE == 2) ((__bf16*)C)[(size_t)m * ldc + n] = (__bf16)r;
    else                ((float*)C)[(size_t)m * ldc + n] = r;
  }
}

// ---------------------------------------------------------------------------
// Attention core: one block per (q-tile of 16, head, batch).
// scores = Q@K^T / 8 -> softmax (f32 math, bf16 probs) -> @V^T, all WMMA.
// AV phase uses split-K across wave pairs (all 8 waves busy).
// ---------------------------------------------------------------------------
__global__ void __launch_bounds__(256) attn_kernel(
    const __bf16* __restrict__ Qb, const __bf16* __restrict__ Kb,
    const __bf16* __restrict__ Vt, float* __restrict__ Oa) {
  int qt = blockIdx.x;   // 0..63
  int h  = blockIdx.y;   // 0..7
  int b  = blockIdx.z;   // 0..1
  int tid = threadIdx.x, lane = tid & 31, wave = tid >> 5;
  int hi = lane >> 4;

  __shared__ __bf16 sq[16 * HDIM];   // 2KB
  __shared__ __bf16 sp[16 * 1024];   // 32KB scores -> probs
  __shared__ float  red[256];
  __shared__ float  sacc[4 * 256];   // 4KB split-K partial tiles
  __shared__ float  smax[16], sinv[16];

  int row0 = b * 1024 + qt * 16;
  int c0 = h * HDIM;

  // stage Q tile (16B chunks), 128 items
  if (tid < 128) {
    int r = tid >> 3, c8 = tid & 7;
    *(v8bf*)(sq + r * HDIM + c8 * 8) =
        *(const v8bf*)(Qb + (size_t)(row0 + r) * HD + c0 + c8 * 8);
  }
  __syncthreads();

  // scores: 64 key tiles across 8 waves
#pragma unroll
  for (int kk = 0; kk < 8; ++kk) {
    int kt = wave + kk * 8;
    v8f acc = {};
#pragma unroll
    for (int k0 = 0; k0 < HDIM; k0 += 32) {
      v16bf a = load_frag_kc(sq + k0, HDIM);
      v16bf bf = load_frag_kc(Kb + (size_t)(b * 1024 + kt * 16) * HD + c0 + k0, HD);
      acc = wmma_bf16(a, bf, acc);
    }
    int n = kt * 16 + (lane & 15);
#pragma unroll
    for (int v = 0; v < 8; ++v)
      sp[(v + hi * 8) * 1024 + n] = (__bf16)(acc[v] * 0.125f);
  }
  __syncthreads();

  // softmax: 16 threads per row, v8bf chunked reads
  {
    int r = tid >> 4, sub = tid & 15;   // 64 elems each
    float mx = -1e30f;
#pragma unroll
    for (int ch = 0; ch < 8; ++ch) {
      v8bf cv = *(const v8bf*)(sp + r * 1024 + sub * 64 + ch * 8);
#pragma unroll
      for (int j = 0; j < 8; ++j) mx = fmaxf(mx, (float)cv[j]);
    }
    red[tid] = mx;
    __syncthreads();
    if (sub == 0) {
      float m2 = -1e30f;
#pragma unroll
      for (int t2 = 0; t2 < 16; ++t2) m2 = fmaxf(m2, red[(r << 4) + t2]);
      smax[r] = m2;
    }
    __syncthreads();
    float s = 0.f;
#pragma unroll
    for (int ch = 0; ch < 8; ++ch) {
      v8bf cv = *(const v8bf*)(sp + r * 1024 + sub * 64 + ch * 8);
#pragma unroll
      for (int j = 0; j < 8; ++j) s += __expf((float)cv[j] - smax[r]);
    }
    red[tid] = s;
    __syncthreads();
    if (sub == 0) {
      float ss = 0.f;
#pragma unroll
      for (int t2 = 0; t2 < 16; ++t2) ss += red[(r << 4) + t2];
      sinv[r] = 1.f / ss;
    }
  }
  __syncthreads();
  // rescale probs to bf16, 16B chunks
#pragma unroll
  for (int ii = 0; ii < 8; ++ii) {
    int i = tid + ii * 256;            // chunk id 0..2047
    int r = i >> 7;
    v8bf cv = *(const v8bf*)(sp + i * 8);
    v8bf ov;
#pragma unroll
    for (int j = 0; j < 8; ++j)
      ov[j] = (__bf16)(__expf((float)cv[j] - smax[r]) * sinv[r]);
    *(v8bf*)(sp + i * 8) = ov;
  }
  __syncthreads();

  // AV with split-K: wave pair (q, q+4) covers K halves; B from V^T
  {
    int q = wave & 3;
    int khalf = (wave >> 2) * 512;
    v8f acc = {};
    for (int k0 = khalf; k0 < khalf + 512; k0 += 32) {
      v16bf a = load_frag_kc(sp + k0, 1024);
      v16bf bf = load_frag_kc(Vt + (size_t)(c0 + q * 16) * RATT + b * 1024 + k0, RATT);
      acc = wmma_bf16(a, bf, acc);
    }
    if (wave >= 4) {
#pragma unroll
      for (int v = 0; v < 8; ++v) sacc[q * 256 + lane * 8 + v] = acc[v];
    }
    __syncthreads();
    if (wave < 4) {
      int n = c0 + q * 16 + (lane & 15);
#pragma unroll
      for (int v = 0; v < 8; ++v)
        Oa[(size_t)(row0 + v + hi * 8) * HD + n] = acc[v] + sacc[q * 256 + lane * 8 + v];
    }
  }
}

// ---------------------------------------------------------------------------
// Layout shuffles
// ---------------------------------------------------------------------------
__global__ void build_xatt(const float* t0, const float* t1, const float* t2,
                           const float* t3, float* xatt) {
  int i = blockIdx.x * 256 + threadIdx.x;
  if (i >= RATT * HD) return;
  int col = i & (HD - 1);
  int row = i >> 9;
  int b = row >> 10, tr = (row >> 8) & 3, s = row & 255;
  const float* src = (tr == 0) ? t0 : (tr == 1) ? t1 : (tr == 2) ? t2 : t3;
  xatt[i] = src[(size_t)((b << 8) + s) * HD + col];
}

__global__ void build_xc(const float* t0, const float* t1, const float* t2,
                         const float* t3, float* xc) {
  int i = blockIdx.x * 256 + threadIdx.x;
  if (i >= NTOK * 4 * HD) return;
  int n = i >> 11;
  int cc = i & (4 * HD - 1);
  int tr = cc >> 9, col = cc & (HD - 1);
  const float* src = (tr == 0) ? t0 : (tr == 1) ? t1 : (tr == 2) ? t2 : t3;
  xc[i] = src[(size_t)n * HD + col];
}

__global__ void final_kernel(const float* __restrict__ comb,
                             const float* __restrict__ attp,
                             float* __restrict__ out) {
  int i = blockIdx.x * 256 + threadIdx.x;
  if (i >= NTOK * HD) return;
  int col = i & (HD - 1);
  int n = i >> 9;
  int b = n >> 8, s = n & 255;
  float a = 0.f;
#pragma unroll
  for (int t = 0; t < 4; ++t)
    a += attp[(size_t)(b * 1024 + t * 256 + s) * HD + col];
  out[i] = comb[i] + 0.25f * a;
}

// ---------------------------------------------------------------------------
// Host launcher
// ---------------------------------------------------------------------------
extern "C" void kernel_launch(void* const* d_in, const int* in_sizes, int n_in,
                              void* d_out, int out_size, void* d_ws, size_t ws_size,
                              hipStream_t stream) {
  (void)in_sizes; (void)n_in; (void)out_size; (void)ws_size;
  const float* x = (const float*)d_in[0];
  const int tierE[4] = {64, 32, 16, 8};
  const float* tW[4][9];
  for (int t = 0; t < 4; ++t)
    for (int j = 0; j < 9; ++j) tW[t][j] = (const float*)d_in[1 + 9 * t + j];
  const float* Wq = (const float*)d_in[37]; const float* bq = (const float*)d_in[38];
  const float* Wk = (const float*)d_in[39]; const float* bk = (const float*)d_in[40];
  const float* Wv = (const float*)d_in[41]; const float* bv = (const float*)d_in[42];
  const float* Wo = (const float*)d_in[43]; const float* bo = (const float*)d_in[44];
  const float* cW1 = (const float*)d_in[45]; const float* cb1 = (const float*)d_in[46];
  const float* cW2 = (const float*)d_in[47]; const float* cb2 = (const float*)d_in[48];

  // Workspace carve-out
  char* w = (char*)d_ws;
  auto alloc = [&](size_t bytes) -> void* {
    void* p = (void*)w;
    w += (bytes + 255) & ~(size_t)255;
    return p;
  };
  int*    cnt  = (int*)alloc(120 * sizeof(int));
  int*    lidx = (int*)alloc(120 * NTOK * sizeof(int));
  float*  lw   = (float*)alloc(120 * NTOK * sizeof(float));
  float*  tout = (float*)alloc((size_t)4 * NTOK * HD * sizeof(float));
  float*  xatt = (float*)alloc((size_t)RATT * HD * sizeof(float));
  __bf16* Qb   = (__bf16*)alloc((size_t)RATT * HD * sizeof(__bf16));
  __bf16* Kb   = (__bf16*)alloc((size_t)RATT * HD * sizeof(__bf16));
  __bf16* Vt   = (__bf16*)alloc((size_t)RATT * HD * sizeof(__bf16));  // transposed
  float*  oatt = (float*)alloc((size_t)RATT * HD * sizeof(float));
  float*  attp = (float*)alloc((size_t)RATT * HD * sizeof(float));
  float*  xc   = (float*)alloc((size_t)NTOK * 4 * HD * sizeof(float));
  float*  c1   = (float*)alloc((size_t)NTOK * IDIM * sizeof(float));
  float*  comb = (float*)alloc((size_t)NTOK * HD * sizeof(float));
  int tierBase[4] = {0, 64, 96, 112};

  // 1) zero counters + tier outputs
  zero_kernel<<<1, 256, 0, stream>>>((float*)cnt, 120);
  {
    int n = 4 * NTOK * HD;
    zero_kernel<<<(n + 255) / 256, 256, 0, stream>>>(tout, n);
  }

  // 2) routing + 3) expert FFN per tier
  for (int t = 0; t < 4; ++t) {
    int E = tierE[t];
    int* cte   = cnt + tierBase[t];
    int* lie   = lidx + (size_t)tierBase[t] * NTOK;
    float* lwe = lw + (size_t)tierBase[t] * NTOK;
    float* toe = tout + (size_t)t * NTOK * HD;
    router_kernel<<<NTOK, 64, 0, stream>>>(x, tW[t][0], E, cte, lie, lwe);
    dim3 grid(E, 32);
    expert_ffn_kernel<<<grid, 256, 0, stream>>>(
        x, tW[t][1], tW[t][2], tW[t][3], tW[t][4], tW[t][5], tW[t][6],
        tW[t][7], tW[t][8], cte, lie, lwe, toe);
  }
  float* ko = tout, *so = tout + (size_t)NTOK * HD;
  float* mo = tout + (size_t)2 * NTOK * HD, *fo = tout + (size_t)3 * NTOK * HD;

  // 4) stack tiers for attention
  {
    int n = RATT * HD;
    build_xatt<<<(n + 255) / 256, 256, 0, stream>>>(ko, so, mo, fo, xatt);
  }

  // 5) Q/K (bf16 row-major) and V (bf16 transposed) projections
  {
    dim3 grid(HD / 128, RATT / 16);
    gemm_bias_kernel<2><<<grid, 256, 0, stream>>>(xatt, HD, Wq, HD, bq, Qb, HD, HD);
    gemm_bias_kernel<2><<<grid, 256, 0, stream>>>(xatt, HD, Wk, HD, bk, Kb, HD, HD);
    gemm_bias_kernel<3><<<grid, 256, 0, stream>>>(xatt, HD, Wv, HD, bv, Vt, RATT, HD);
  }

  // 6) attention core
  {
    dim3 grid(64, NHEAD, BB);
    attn_kernel<<<grid, 256, 0, stream>>>(Qb, Kb, Vt, oatt);
  }

  // 7) output projection
  {
    dim3 grid(HD / 128, RATT / 16);
    gemm_bias_kernel<0><<<grid, 256, 0, stream>>>(oatt, HD, Wo, HD, bo, attp, HD, HD);
  }

  // 8) combiner MLP on concatenated tiers
  {
    int n = NTOK * 4 * HD;
    build_xc<<<(n + 255) / 256, 256, 0, stream>>>(ko, so, mo, fo, xc);
  }
  {
    dim3 grid(IDIM / 128, NTOK / 16);
    gemm_bias_kernel<1><<<grid, 256, 0, stream>>>(xc, 4 * HD, cW1, IDIM, cb1, c1, IDIM, 4 * HD);
  }
  {
    dim3 grid(HD / 128, NTOK / 16);
    gemm_bias_kernel<0><<<grid, 256, 0, stream>>>(c1, IDIM, cW2, HD, cb2, comb, HD, IDIM);
  }

  // 9) final: comb + mean-over-tiers attention
  {
    int n = NTOK * HD;
    final_kernel<<<(n + 255) / 256, 256, 0, stream>>>(comb, attp, (float*)d_out);
  }
}
